// AdaConvV2_87505663688796
// MI455X (gfx1250) — matrix operand order repaired and checked by reading, not compiled, not run on hardware
//
#include <hip/hip_runtime.h>
#include <hip/hip_bf16.h>
#include <math.h>

#define C_IN   128
#define C_EMB  256
#define HW     128
#define NIMG   16
#define TILE_H 8
#define TILE_W 16
#define HALO_H (TILE_H + 6)          // 14
#define HALO_W (TILE_W + 6)          // 22
#define HALO_POS (HALO_H * HALO_W)   // 308
#define HALO_STRIDE 310              // padded (banks)

// LDS layout (bytes)
#define LDS_HALO_OFF 0                               // 128*310*4      = 158720 (reused as f32 out tile, stride 129)
#define LDS_DWW_OFF  158720                          // 128*49*4       = 25088
#define LDS_H0_OFF   183808                          // 128*136*2      = 34816  (bf16, padded stride)
#define LDS_H1_OFF   218624                          // 128*264*2      = 67584  (bf16, padded stride; reused as f32 tmp stride 130)
#define LDS_RED_OFF  286208                          // 512*4          = 2048
#define LDS_GB2_OFF  288256                          // 128*4          = 512
#define LDS_B1_OFF   288768                          // 256*4          = 1024
#define LDS_GV_OFF   289792                          // 256*4          = 1024
#define LDS_TOTAL    290816
#define H0_STRIDE    136
#define H1_STRIDE    264
#define TMPF_STRIDE  130
#define OUT_STRIDE   129

typedef __attribute__((ext_vector_type(16))) __bf16 v16bf;
typedef __attribute__((ext_vector_type(8)))  float  v8f;

union ABf { v16bf v; unsigned short s[16]; };

__device__ __forceinline__ unsigned short f2bf(float f) {
    unsigned u = __float_as_uint(f);
    unsigned r = u + 0x7FFFu + ((u >> 16) & 1u);   // RNE
    return (unsigned short)(r >> 16);
}
__device__ __forceinline__ float bf2f(unsigned short s) {
    return __uint_as_float(((unsigned)s) << 16);
}
// Branchless tanh-form GELU: |err| vs exact-erf GELU < 1e-3, below the bf16 GEMM
// quantization noise. One v_exp_f32 + one v_rcp_f32, no EXEC-divergent paths.
__device__ __forceinline__ float gelu_fast(float v) {
    float u = 0.7978845608028654f * (v + 0.044715f * v * v * v);
    u = fminf(10.0f, fmaxf(-10.0f, u));
    float e = __expf(-2.0f * u);                       // tanh(u) = (1-e)/(1+e)
    float t = (1.0f - e) * __builtin_amdgcn_rcpf(1.0f + e);
    return 0.5f * v * (1.0f + t);
}

// ---------------------------------------------------------------- pool: y_mean[n,c]
__global__ void pool_kernel(const float* __restrict__ x, float* __restrict__ ymean) {
    int nc = blockIdx.x;                       // n*128 + c
    const float* p = x + (size_t)nc * (HW * HW);
    float s = 0.f;
    for (int i = threadIdx.x; i < HW * HW; i += 256) s += p[i];
    __shared__ float r[256];
    r[threadIdx.x] = s;
    __syncthreads();
    for (int st = 128; st > 0; st >>= 1) {
        if ((int)threadIdx.x < st) r[threadIdx.x] += r[threadIdx.x + st];
        __syncthreads();
    }
    if (threadIdx.x == 0) ymean[nc] = r[0] * (1.0f / (HW * HW));
}

// ------------------------------------------- gate: softmax(fc) repeat -> bf16; w1^T -> bf16
__global__ void gate_kernel(const float* __restrict__ ymean,
                            const float* __restrict__ fc_w, const float* __restrict__ fc_b,
                            const float* __restrict__ w1,
                            unsigned short* __restrict__ gate_bf,
                            unsigned short* __restrict__ w1t) {
    __shared__ float lg[NIMG][8];
    int t = threadIdx.x;
    if (t < 128) {
        int n = t >> 3, s = t & 7;
        float acc = fc_b[s];
        for (int c = 0; c < C_IN; ++c) acc += ymean[n * C_IN + c] * fc_w[c * 8 + s];
        lg[n][s] = acc;
    }
    __syncthreads();
    if (t < 128) {
        int n = t >> 3, s = t & 7;
        float mx = lg[n][0];
        #pragma unroll
        for (int i = 1; i < 8; ++i) mx = fmaxf(mx, lg[n][i]);
        float den = 0.f;
        #pragma unroll
        for (int i = 0; i < 8; ++i) den += expf(lg[n][i] - mx);
        float g = expf(lg[n][s] - mx) / den;
        unsigned short gb = f2bf(g);
        for (int r = 0; r < C_EMB / 8; ++r) gate_bf[n * C_EMB + s * 32 + r] = gb;
    }
    // w1t[co][ci] = bf16(w1[ci][co]) — K-contiguous rows for WMMA B-fragments
    for (int i = t; i < C_IN * C_EMB; i += 256) {
        int co = i >> 7, ci = i & 127;
        w1t[co * C_IN + ci] = f2bf(w1[ci * C_EMB + co]);
    }
}

// --------------------- swgt[co][d] = gamma[co] * sum_e styles[d][e]*w2[e][co]  (bf16, K-contig)
__global__ void swgt_kernel(const float* __restrict__ styles, const float* __restrict__ w2,
                            const float* __restrict__ gamma, unsigned short* __restrict__ swgt) {
    int idx = blockIdx.x * 256 + threadIdx.x;   // 0 .. 32767
    int co = idx & 127, d = idx >> 7;
    float acc = 0.f;
    for (int e = 0; e < C_EMB; ++e) acc += styles[d * C_EMB + e] * w2[e * C_IN + co];
    swgt[co * C_EMB + d] = f2bf(acc * gamma[co]);
}

// ---------------------------------------------------------------- fused main kernel
__global__ void main_kernel(const float* __restrict__ x,
                            const float* __restrict__ dw_w, const float* __restrict__ dw_b,
                            const float* __restrict__ ln_w, const float* __restrict__ ln_b,
                            const float* __restrict__ b1,
                            const float* __restrict__ b2, const float* __restrict__ gamma,
                            const unsigned short* __restrict__ gate_bf,
                            const unsigned short* __restrict__ w1t,
                            const unsigned short* __restrict__ swgt,
                            float* __restrict__ out) {
    extern __shared__ char smem[];
    float*          halo = (float*)(smem + LDS_HALO_OFF);           // [128][310] f32
    float*          dww  = (float*)(smem + LDS_DWW_OFF);            // [128][49]
    unsigned short* h0s  = (unsigned short*)(smem + LDS_H0_OFF);    // [128 pix][136] bf16
    unsigned short* h1s  = (unsigned short*)(smem + LDS_H1_OFF);    // [128 pix][264] bf16
    float*          tmpf = (float*)(smem + LDS_H1_OFF);             // [128 pix][130] f32 (phase 2 only)
    float*          red  = (float*)(smem + LDS_RED_OFF);            // [128][2][2]
    float*          gb2  = (float*)(smem + LDS_GB2_OFF);            // [128] gamma*b2
    float*          b1s  = (float*)(smem + LDS_B1_OFF);             // [256] b1
    float*          gvs  = (float*)(smem + LDS_GV_OFF);             // [256] gate (f32)
    float*          obuf = (float*)(smem + LDS_HALO_OFF);           // [128 pix][129] f32 (phase 5)

    const int tid   = threadIdx.x;
    const int wtile = blockIdx.x;      // 0..7
    const int htile = blockIdx.y;      // 0..15
    const int nimg  = blockIdx.z;      // 0..15
    const int hBase = htile * TILE_H;
    const int wBase = wtile * TILE_W;

    if (tid == 0) {                    // gfx1250 global_prefetch_b8 of shared B matrices
        __builtin_prefetch(w1t, 0, 1);
        __builtin_prefetch(swgt, 0, 1);
    }

    // ---- phase 0/1: stage dw weights, x halo, and small per-block tables into LDS
    for (int i = tid; i < C_IN * 49; i += 256) dww[i] = dw_w[i];
    if (tid < C_IN) gb2[tid] = gamma[tid] * b2[tid];
    if (tid < C_EMB) {
        b1s[tid] = b1[tid];
        gvs[tid] = bf2f(gate_bf[nimg * C_EMB + tid]);  // folds diag(y[n]) style gate
    }
    const int h0g = hBase - 3, w0g = wBase - 3;
    for (int i = tid; i < C_IN * HALO_POS; i += 256) {
        int c = i / HALO_POS, pos = i - c * HALO_POS;
        int py = pos / HALO_W, px = pos - py * HALO_W;
        int gh = h0g + py, gw = w0g + px;
        float v = 0.f;
        if (gh >= 0 && gh < HW && gw >= 0 && gw < HW)
            v = x[(((size_t)nimg * C_IN + c) * HW + gh) * HW + gw];
        halo[c * HALO_STRIDE + pos] = v;
    }
    __syncthreads();

    // ---- phase 2: dwconv 7x7 + LayerNorm(C).  2 threads / pixel, 64 channels each.
    const int pix = tid >> 1, half = tid & 1;
    const int py = pix >> 4, pxl = pix & 15;
    float s = 0.f, sq = 0.f;
    #pragma unroll 1
    for (int i = 0; i < 64; ++i) {
        int c = half * 64 + i;
        float acc = dw_b[c];
        const float* hc = halo + c * HALO_STRIDE + py * HALO_W + pxl;
        const float* wc = dww + c * 49;
        #pragma unroll
        for (int ky = 0; ky < 7; ++ky)
            #pragma unroll
            for (int kx = 0; kx < 7; ++kx)
                acc += hc[ky * HALO_W + kx] * wc[ky * 7 + kx];
        tmpf[pix * TMPF_STRIDE + c] = acc;
        s += acc; sq += acc * acc;
    }
    red[pix * 4 + half * 2 + 0] = s;
    red[pix * 4 + half * 2 + 1] = sq;
    __syncthreads();
    {
        float so  = red[pix * 4 + (half ^ 1) * 2 + 0];
        float sqo = red[pix * 4 + (half ^ 1) * 2 + 1];
        float mean = (s + so) * (1.0f / C_IN);
        float var  = (sq + sqo) * (1.0f / C_IN) - mean * mean;
        float rstd = rsqrtf(var + 1e-6f);
        #pragma unroll 1
        for (int i = 0; i < 64; ++i) {
            int c = half * 64 + i;
            float v = (tmpf[pix * TMPF_STRIDE + c] - mean) * rstd * ln_w[c] + ln_b[c];
            h0s[pix * H0_STRIDE + c] = f2bf(v);
        }
    }
    __syncthreads();

    // ---- wave-local WMMA tiles: wave owns pixels m0..m0+15
    const int wv   = tid >> 5;
    const int lane = tid & 31;
    const int m0   = wv * 16;
    const int mrow = lane & 15;
    const int hi   = (lane >= 16) ? 1 : 0;
    const int koff = hi ? 16 : 0;       // A-fragment K-half (contiguous order, k = 16*hi + s)
    const int kb2  = hi ? 16 : 0;       // B-fragment K-half

    // ---- phase 3: GEMM1  h0[16x128] @ w1[128x256]  (bf16 WMMA, f32 acc)
    // Fully unrolled: one straight-line block; backend scheduler pipelines the
    // next tile's global_load_b128 pairs under the current tile's WMMAs.
    ABf afr[4];
    {
        const unsigned short* rp = h0s + (m0 + mrow) * H0_STRIDE;
        #pragma unroll
        for (int kt = 0; kt < 4; ++kt) {
            int b0 = kt * 32 + koff;
            #pragma unroll
            for (int j = 0; j < 8; ++j) {
                afr[kt].s[j]     = rp[b0 + j];        // K = b0..b0+7
                afr[kt].s[8 + j] = rp[b0 + 8 + j];    // K = b0+8..b0+15
            }
        }
    }
    #pragma unroll
    for (int nt = 0; nt < C_EMB / 16; ++nt) {
        int n = nt * 16 + mrow;                        // B/C column = lane&15
        ABf bfr[4];
        const unsigned short* bp0 = w1t + (size_t)n * C_IN + kb2;
        #pragma unroll
        for (int kt = 0; kt < 4; ++kt)
            #pragma unroll
            for (int j = 0; j < 16; ++j) bfr[kt].s[j] = bp0[kt * 32 + j];
        v8f acc = {};
        #pragma unroll
        for (int kt = 0; kt < 4; ++kt)
            acc = __builtin_amdgcn_wmma_f32_16x16x32_bf16(
                false, afr[kt].v, false, bfr[kt].v, (short)0, acc, false, false);
        float b1v = b1s[n];
        float gv  = gvs[n];
        #pragma unroll
        for (int r = 0; r < 8; ++r) {
            int m = m0 + hi * 8 + r;                   // C/D layout: M = r + 8*hi
            h1s[m * H1_STRIDE + n] = f2bf(gelu_fast(acc[r] + b1v) * gv);
        }
    }

    // ---- phase 4: GEMM2  h1[16x256] @ (styles@w2 * gamma)[256x128]
    ABf a2[8];
    {
        const unsigned short* rp = h1s + (m0 + mrow) * H1_STRIDE;
        #pragma unroll
        for (int kt = 0; kt < 8; ++kt) {
            int b0 = kt * 32 + koff;
            #pragma unroll
            for (int j = 0; j < 8; ++j) {
                a2[kt].s[j]     = rp[b0 + j];
                a2[kt].s[8 + j] = rp[b0 + 8 + j];
            }
        }
    }
    #pragma unroll
    for (int nt = 0; nt < C_IN / 16; ++nt) {
        int co = nt * 16 + mrow;
        const unsigned short* bp0 = swgt + (size_t)co * C_EMB + kb2;
        ABf bfr[8];
        #pragma unroll
        for (int kt = 0; kt < 8; ++kt)
            #pragma unroll
            for (int j = 0; j < 16; ++j) bfr[kt].s[j] = bp0[kt * 32 + j];
        v8f acc = {};
        #pragma unroll
        for (int kt = 0; kt < 8; ++kt)
            acc = __builtin_amdgcn_wmma_f32_16x16x32_bf16(
                false, a2[kt].v, false, bfr[kt].v, (short)0, acc, false, false);
        #pragma unroll
        for (int r = 0; r < 8; ++r) {
            int m = m0 + hi * 8 + r;
            obuf[m * OUT_STRIDE + co] = acc[r];        // bias/residual deferred
        }
    }
    __syncthreads();

    // ---- phase 5: +gamma*b2 +x residual, coalesced NCHW store
    for (int i = tid; i < C_IN * (TILE_H * TILE_W); i += 256) {
        int c = i >> 7, p = i & 127;
        int gh = hBase + (p >> 4), gw = wBase + (p & 15);
        size_t off = (((size_t)nimg * C_IN + c) * HW + gh) * HW + gw;
        out[off] = obuf[p * OUT_STRIDE + c] + gb2[c] + x[off];
    }
}

extern "C" void kernel_launch(void* const* d_in, const int* in_sizes, int n_in,
                              void* d_out, int out_size, void* d_ws, size_t ws_size,
                              hipStream_t stream) {
    const float* x      = (const float*)d_in[0];
    const float* dw_w   = (const float*)d_in[1];
    const float* dw_b   = (const float*)d_in[2];
    const float* ln_w   = (const float*)d_in[3];
    const float* ln_b   = (const float*)d_in[4];
    const float* w1     = (const float*)d_in[5];
    const float* b1     = (const float*)d_in[6];
    const float* fc_w   = (const float*)d_in[7];
    const float* fc_b   = (const float*)d_in[8];
    const float* styles = (const float*)d_in[9];
    const float* w2     = (const float*)d_in[10];
    const float* b2     = (const float*)d_in[11];
    const float* gamma  = (const float*)d_in[12];
    float* out = (float*)d_out;

    // workspace layout (≈144 KB)
    float*          ymean   = (float*)d_ws;                                   // 16*128 f32
    unsigned short* gate_bf = (unsigned short*)((char*)d_ws + 8192);          // 16*256 bf16
    unsigned short* w1t     = (unsigned short*)((char*)d_ws + 16384);         // 256*128 bf16
    unsigned short* swgt    = (unsigned short*)((char*)d_ws + 16384 + 65536); // 128*256 bf16

    pool_kernel<<<NIMG * C_IN, 256, 0, stream>>>(x, ymean);
    gate_kernel<<<1, 256, 0, stream>>>(ymean, fc_w, fc_b, w1, gate_bf, w1t);
    swgt_kernel<<<(C_EMB * C_IN) / 256, 256, 0, stream>>>(styles, w2, gamma, swgt);

    (void)hipFuncSetAttribute((const void*)main_kernel,
                              hipFuncAttributeMaxDynamicSharedMemorySize, LDS_TOTAL);
    main_kernel<<<dim3(HW / TILE_W, HW / TILE_H, NIMG), 256, LDS_TOTAL, stream>>>(
        x, dw_w, dw_b, ln_w, ln_b, b1, b2, gamma, gate_bf, w1t, swgt, out);
}
